// BOCPD_GPTS_16174846837166
// MI455X (gfx1250) — compile-verified, matrix operand order checked by reading, not run on hardware
//
#include <hip/hip_runtime.h>
#include <math.h>

// BOCPD-GPTS on MI455X (gfx1250, wave32).
// Z[t] == upm[t] for constant hazard (r stays normalized, tail stays 0), so
// all 4096 steps are independent. One wave32 workgroup per step:
//   - 128x128 padded RBF kernel matrix in LDS (exactly 64 KB),
//   - blocked right-looking Cholesky, 16-wide panels,
//     trailing SYRK tiles on the WMMA pipe (f32 16x16x4 if available,
//     else confirmed f32<-f16 16x16x32 with zero-padded K),
//   - two forward substitutions with register-resident RHS/solution,
//   - StudentT log-pdf -> out[t-1].

typedef __attribute__((ext_vector_type(2)))  float    v2f;
typedef __attribute__((ext_vector_type(8)))  float    v8f;
typedef __attribute__((ext_vector_type(16))) _Float16 v16h;

#define TLEN 4096
#define WWIN 128
#define MPAD 128   // real window M=127, padded with one identity row/col
#define NB   8     // 128 / 16 panels

// One 16x16 SYRK trailing-update tile: C -= P_bi * P_bj^T, K=16 panel at c0.
// Lane mapping (wave32): nn = lane&15, hf = lane>>4.
__device__ __forceinline__ void syrk_tile(float* Kf, int c0, int r0, int q0, int l)
{
    const int nn = l & 15;
    const int hf = l >> 4;
    v8f c;
    #pragma unroll
    for (int v = 0; v < 8; ++v)
        c[v] = Kf[(r0 + v + 8 * hf) * MPAD + q0 + nn];

#if __has_builtin(__builtin_amdgcn_wmma_f32_16x16x4_f32)
    // Native f32 path: 4 chained K=4 WMMAs. A negated -> C -= A*B.
    #pragma unroll
    for (int q = 0; q < 4; ++q) {
        int t0 = c0 + 4 * q;
        v2f a, b;
        a.x = -Kf[(r0 + nn) * MPAD + t0 + 2 * hf];
        a.y = -Kf[(r0 + nn) * MPAD + t0 + 2 * hf + 1];
        b.x =  Kf[(q0 + nn) * MPAD + t0 + 2 * hf];
        b.y =  Kf[(q0 + nn) * MPAD + t0 + 2 * hf + 1];
        c = __builtin_amdgcn_wmma_f32_16x16x4_f32(
                false, a, false, b, (short)0, c, false, false);
    }
#elif __has_builtin(__builtin_amdgcn_wmma_f32_16x16x32_f16)
    // Confirmed f16-input path: zero-pad K=16 -> 32, accumulate f32.
    // A (16x32 f16) lane layout: half m<8 <-> K = 8*hf + m; m>=8 <-> K>=16 (pad 0).
    // B (32x16 f16) lane layout: lanes 0-15 hold K=0-15, lanes 16-31 K=16-31 (pad 0).
    v16h a, b;
    #pragma unroll
    for (int m = 0; m < 16; ++m)
        a[m] = (m < 8) ? (_Float16)(-Kf[(r0 + nn) * MPAD + c0 + 8 * hf + m])
                       : (_Float16)0.0f;
    #pragma unroll
    for (int m = 0; m < 16; ++m) {
        int kk = 16 * hf + m;
        b[m] = (kk < 16) ? (_Float16)Kf[(q0 + nn) * MPAD + c0 + kk]
                         : (_Float16)0.0f;
    }
    c = __builtin_amdgcn_wmma_f32_16x16x32_f16(
            false, a, false, b, (short)0, c, false, false);
#else
    // No matrix builtin visible in this pass (e.g. host stub): keep C as-is.
    (void)c0;
#endif

    #pragma unroll
    for (int v = 0; v < 8; ++v)
        Kf[(r0 + v + 8 * hf) * MPAD + q0 + nn] = c[v];
}

__global__ __launch_bounds__(32)
void bocpd_step_kernel(const float* __restrict__ X, const float* __restrict__ Y,
                       const float* __restrict__ log_noise,
                       const float* __restrict__ log_amp,
                       const float* __restrict__ log_ls,
                       float* __restrict__ out)
{
    __shared__ float Kf[MPAD * MPAD];   // 64 KB, row-major, stride 128
    const int l = threadIdx.x;          // 0..31 (wave32)
    const int t = blockIdx.x + 1;       // 1..4096

    const float noise = __expf(log_noise[0]);
    const float amp   = __expf(log_amp[0]);
    const float ls2   = __expf(2.0f * log_ls[0]);

    int start = t - WWIN; if (start < 0) start = 0;
    int nv = (t >= WWIN) ? (WWIN - 1) : ((t - 1) < 1 ? 1 : (t - 1));
    const float xt = X[t - 1];

    // window X/Y and Kx kept register-resident: index i = s*32 + lane
    float xs[4], ys[4], kx[4];
    #pragma unroll
    for (int s = 0; s < 4; ++s) {
        int i = s * 32 + l;
        float xv = (i < WWIN - 1) ? X[start + i] : 0.0f;
        bool val = (i < nv);
        xs[s] = xv;
        ys[s] = val ? Y[start + i] : 0.0f;
        float d = xv - xt;
        kx[s] = val ? amp * __expf(-0.5f * d * d / ls2) : 0.0f;
    }

    // ---- build Kf = [kern + noise*I on valid block ; identity pad] ----
    #pragma unroll
    for (int shi = 0; shi < 4; ++shi) {
        for (int ii = 0; ii < 32; ++ii) {
            int i = shi * 32 + ii;
            float xi = __shfl(xs[shi], ii, 32);   // broadcast row point
            bool vi = (i < nv);
            #pragma unroll
            for (int sub = 0; sub < 4; ++sub) {
                int j = sub * 32 + l;
                float v;
                if (vi && (j < nv)) {
                    float d = xi - xs[sub];
                    v = amp * __expf(-0.5f * d * d / ls2) + ((i == j) ? noise : 0.0f);
                } else {
                    v = (i == j) ? 1.0f : 0.0f;
                }
                Kf[i * MPAD + j] = v;
            }
        }
    }
    __syncthreads();

    // ---- blocked right-looking Cholesky (in-place, lower triangle) ----
    for (int k = 0; k < NB; ++k) {
        const int c0 = 16 * k;

        // panel factorization: columns c0..c0+15, all rows below
        for (int cl = 0; cl < 16; ++cl) {
            int c = c0 + cl;
            float diag = 0.0f;                         // redundant per lane
            for (int tt = c0; tt < c; ++tt) {
                float lv = Kf[c * MPAD + tt];
                diag += lv * lv;
            }
            float Ld = sqrtf(Kf[c * MPAD + c] - diag);
            for (int r = c + 1 + l; r < MPAD; r += 32) {
                float acc = 0.0f;
                for (int tt = c0; tt < c; ++tt)
                    acc += Kf[r * MPAD + tt] * Kf[c * MPAD + tt];
                Kf[r * MPAD + c] = (Kf[r * MPAD + c] - acc) / Ld;
            }
            if (l == 0) Kf[c * MPAD + c] = Ld;
            __syncthreads();
        }

        // trailing SYRK: C_{bi,bj} -= P_bi * P_bj^T (wave-uniform loops)
        for (int bi = k + 1; bi < NB; ++bi)
            for (int bj = k + 1; bj <= bi; ++bj)
                syrk_tile(Kf, c0, 16 * bi, 16 * bj, l);
        __syncthreads();
    }

    // ---- forward substitution: A = L^-1 Kx, V = L^-1 Yw (register RHS) ----
    float sa[4], sv[4];
    #pragma unroll
    for (int s = 0; s < 4; ++s) { sa[s] = 0.0f; sv[s] = 0.0f; }
    #pragma unroll
    for (int s = 0; s < 4; ++s) {
        for (int cc = 0; cc < 32; ++cc) {
            int c = s * 32 + cc;
            float accA = 0.0f, accV = 0.0f;
            #pragma unroll
            for (int ss = 0; ss < 4; ++ss) {
                int tt = ss * 32 + l;
                if (tt < c) {
                    float lv = Kf[c * MPAD + tt];
                    accA += lv * sa[ss];
                    accV += lv * sv[ss];
                }
            }
            #pragma unroll
            for (int off = 16; off > 0; off >>= 1) {
                accA += __shfl_xor(accA, off, 32);
                accV += __shfl_xor(accV, off, 32);
            }
            float Ld = Kf[c * MPAD + c];
            if (l == cc) {
                sa[s] = (kx[s] - accA) / Ld;
                sv[s] = (ys[s] - accV) / Ld;
            }
        }
    }

    float mu = 0.0f, var = 0.0f;
    #pragma unroll
    for (int s = 0; s < 4; ++s) { mu += sa[s] * sa[s]; var += sv[s] * sv[s]; }
    #pragma unroll
    for (int off = 16; off > 0; off >>= 1) {
        mu  += __shfl_xor(mu,  off, 32);
        var += __shfl_xor(var, off, 32);
    }

    if (l == 0) {
        float df = (float)(t + 1);
        float y  = Y[t - 1];
        float z  = (y - mu) / var;
        float lp = lgammaf(0.5f * (df + 1.0f)) - lgammaf(0.5f * df)
                 - 0.5f * logf(df * 3.14159265358979323846f) - logf(var)
                 - 0.5f * (df + 1.0f) * log1pf(z * z / df);
        out[blockIdx.x] = __expf(lp);   // Z[t] == upm[t] (constant hazard)
    }
}

extern "C" void kernel_launch(void* const* d_in, const int* in_sizes, int n_in,
                              void* d_out, int out_size, void* d_ws, size_t ws_size,
                              hipStream_t stream)
{
    (void)in_sizes; (void)n_in; (void)out_size; (void)d_ws; (void)ws_size;
    const float* X  = (const float*)d_in[0];
    const float* Y  = (const float*)d_in[1];
    const float* ln = (const float*)d_in[2];
    const float* la = (const float*)d_in[3];
    const float* ll = (const float*)d_in[4];
    float* out = (float*)d_out;
    bocpd_step_kernel<<<TLEN, 32, 0, stream>>>(X, Y, ln, la, ll, out);
}
